// MinkUNetDiff_43499428774433
// MI455X (gfx1250) — matrix operand order repaired and checked by reading, not compiled, not run on hardware
//
#include <hip/hip_runtime.h>
#include <hip/hip_bf16.h>

// ============================================================================
// MinkUNet diffusion forward for MI455X (gfx1250, wave32, WMMA).
//
// Core kernel: fused gather -> bf16 WMMA GEMM (f32 accumulate) -> scatter,
// one launch per rulebook offset. Scatter targets are unique per offset, so
// plain store / read-modify-write, no atomics -> deterministic.
//
// Weights are pre-packed once per layer into bf16 WMMA B-fragments
// (32 bytes per lane per fragment) so the GEMM inner loop does one aligned
// 32B vector load per WMMA instead of 16 column-strided f32 scalar loads.
//
// Layout (CDNA5 ISA):
//   A (16x32 bf16): lane l: g=l>>4, m=l&15; VGPR j holds K = (j<4 ? 2j : 16+2(j-4)) + 8g
//   B (32x16 bf16): lane l: g=l>>4, n=l&15; VGPR j holds K = (j<4 ? 2j : 16+2(j-4)) + 8g   (column n)
//   D (16x16 f32) : lane l: n=l&15; VGPR r holds row m = r + 8*(l>>4)
//
// Output-coverage modes (avoids zero-fill + RMW traffic):
//   k=3 convs : center offset (13) is identity over all rows -> store first,
//               other 26 offsets accumulate.
//   up convs  : the 8 offsets partition output rows -> every launch stores.
//   down convs: many-to-one -> zero destination, all launches accumulate.
//
// Input flattening assumption: dict-insertion-order recursive flatten of
// setup_inputs(); n_l recovered from len(k3[l].ins[13]) (center identity map).
// ============================================================================

typedef __attribute__((ext_vector_type(16))) __bf16          v16bf;
typedef __attribute__((ext_vector_type(16))) unsigned short  v16u;
typedef __attribute__((ext_vector_type(8)))  float           v8f;

__device__ __forceinline__ unsigned short f2bf(float f) {
  unsigned u = __builtin_bit_cast(unsigned, f);
  u += 0x7fffu + ((u >> 16) & 1u);            // round-to-nearest-even
  return (unsigned short)(u >> 16);
}

// ---------------------------------------------------------------------------
// Pack f32 weights (nk, Cin, Cout) into bf16 WMMA B-fragments.
// Fragment index: ((ko*KC + kc)*CT + ct), 32 lanes x 16 bf16 each.
// Zero-padded for kb >= Cin and col >= Cout, so the GEMM needs no B guards.
// grid = (KC, CT, nk), block = 32.
// ---------------------------------------------------------------------------
__global__ __launch_bounds__(32) void k_pack_w(
    const float* __restrict__ w, int Cin, int Cout,
    unsigned short* __restrict__ pack)
{
  const int kc = blockIdx.x, ct = blockIdx.y, ko = blockIdx.z;
  const int KC = gridDim.x, CT = gridDim.y;
  const int lane = threadIdx.x;
  const int g = lane >> 4, nr = lane & 15;
  const int col = ct * 16 + nr;
  const float* wk = w + (size_t)ko * Cin * Cout;
  unsigned short* dst = pack + ((((size_t)ko * KC + kc) * CT + ct) * 32 + lane) * 16;
#pragma unroll
  for (int j = 0; j < 8; ++j) {
    int kb = kc * 32 + ((j < 4) ? 2 * j : 16 + 2 * (j - 4)) + 8 * g;
    float f0 = (col < Cout && kb     < Cin) ? wk[(long)kb       * Cout + col] : 0.f;
    float f1 = (col < Cout && kb + 1 < Cin) ? wk[(long)(kb + 1) * Cout + col] : 0.f;
    dst[2 * j]     = f2bf(f0);
    dst[2 * j + 1] = f2bf(f1);
  }
}

// ---------------------------------------------------------------------------
// y[outs[m], :] (+)= x[ins[m], :] @ W      (W given as packed bf16 fragments)
// ins/outs == nullptr => identity. One wave per block; 16 x 64 output tile.
// ---------------------------------------------------------------------------
__global__ __launch_bounds__(32) void k_spconv_wmma(
    const float* __restrict__ x, int ldx,
    const unsigned short* __restrict__ wp,   // fragments for this offset
    const int* __restrict__ ins, const int* __restrict__ outs,
    int M, float* __restrict__ y, int ldy, int Cin, int Cout,
    int ctiles, int store)
{
  const int lane = threadIdx.x;
  const int g  = lane >> 4;
  const int mr = lane & 15;
  const int m0 = blockIdx.x * 16;
  const int nb = blockIdx.y * 64;
  const int nct = min(4, (Cout - nb + 15) >> 4);

  const int pair = m0 + mr;
  long row = -1;
  if (pair < M) row = ins ? (long)ins[pair] : (long)pair;

  v8f acc[4];
#pragma unroll
  for (int c = 0; c < 4; ++c)
#pragma unroll
    for (int r = 0; r < 8; ++r) acc[c][r] = 0.f;

  const int KC = (Cin + 31) >> 5;
  const bool aligned = (ldx & 3) == 0;

  for (int kc = 0; kc < KC; ++kc) {
    const int k0 = kc << 5;
    // ---- gather + convert A tile (16x32 bf16) ----
    v16u au;
    if (row >= 0) {
      if (aligned && (k0 + 32 <= Cin)) {
        const float4* xp = (const float4*)(x + row * (long)ldx + k0);
        float4 p0 = xp[2 * g];
        float4 p1 = xp[2 * g + 1];
        float4 p2 = xp[2 * g + 4];
        float4 p3 = xp[2 * g + 5];
        au[0]  = f2bf(p0.x); au[1]  = f2bf(p0.y); au[2]  = f2bf(p0.z); au[3]  = f2bf(p0.w);
        au[4]  = f2bf(p1.x); au[5]  = f2bf(p1.y); au[6]  = f2bf(p1.z); au[7]  = f2bf(p1.w);
        au[8]  = f2bf(p2.x); au[9]  = f2bf(p2.y); au[10] = f2bf(p2.z); au[11] = f2bf(p2.w);
        au[12] = f2bf(p3.x); au[13] = f2bf(p3.y); au[14] = f2bf(p3.z); au[15] = f2bf(p3.w);
      } else {
#pragma unroll
        for (int j = 0; j < 8; ++j) {
          int kb = k0 + ((j < 4) ? 2 * j : 16 + 2 * (j - 4)) + 8 * g;
          float f0 = (kb     < Cin) ? x[row * (long)ldx + kb    ] : 0.f;
          float f1 = (kb + 1 < Cin) ? x[row * (long)ldx + kb + 1] : 0.f;
          au[2 * j]     = f2bf(f0);
          au[2 * j + 1] = f2bf(f1);
        }
      }
    } else {
#pragma unroll
      for (int e = 0; e < 16; ++e) au[e] = 0;
    }
    v16bf a = __builtin_bit_cast(v16bf, au);

    // ---- 4 WMMAs against pre-packed B fragments ----
    const v16u* bbase = (const v16u*)wp + (size_t)kc * ctiles * 32 + lane;
#pragma unroll
    for (int c = 0; c < 4; ++c) {
      if (c >= nct) break;                     // uniform across the wave
      int ctg = (nb >> 4) + c;
      v16u bu = bbase[(size_t)ctg * 32];
      v16bf b = __builtin_bit_cast(v16bf, bu);
      acc[c] = __builtin_amdgcn_wmma_f32_16x16x32_bf16(
          /*neg_a=*/false, a, /*neg_b=*/false, b,
          /*c_mod=*/(short)0, acc[c], /*reuse_a=*/false, /*reuse_b=*/false);
    }
  }

  // ---- scatter D (unique rows within this launch) ----
#pragma unroll
  for (int c = 0; c < 4; ++c) {
    if (c >= nct) break;
    int col = nb + c * 16 + mr;
    if (col >= Cout) continue;
#pragma unroll
    for (int r = 0; r < 8; ++r) {
      int pm = m0 + 8 * g + r;
      if (pm < M) {
        long orow = outs ? (long)outs[pm] : (long)pm;
        if (store) y[orow * (long)ldy + col]  = acc[c][r];
        else       y[orow * (long)ldy + col] += acc[c][r];
      }
    }
  }
}

// ---------------------------------------------------------------------------
// Elementwise / BN / embedding kernels
// ---------------------------------------------------------------------------
__global__ void k_zero(float* p, long n) {
  long i = (long)blockIdx.x * blockDim.x + threadIdx.x;
  if (i < n) p[i] = 0.f;
}

// one block per channel -> deterministic (no atomics); cross-channel blocks
// sweep the same rows concurrently, so lines are fetched once into L2.
__global__ void k_bn_stats(const float* __restrict__ x, int ld, int n, int C,
                           float* __restrict__ stats) {
  __shared__ float ss[256], sq[256];
  int c = blockIdx.x;
  float s = 0.f, q = 0.f;
  for (long i = threadIdx.x; i < n; i += blockDim.x) {
    float v = x[i * (long)ld + c];
    s += v; q += v * v;
  }
  ss[threadIdx.x] = s; sq[threadIdx.x] = q;
  __syncthreads();
  for (int o = 128; o > 0; o >>= 1) {
    if ((int)threadIdx.x < o) {
      ss[threadIdx.x] += ss[threadIdx.x + o];
      sq[threadIdx.x] += sq[threadIdx.x + o];
    }
    __syncthreads();
  }
  if (threadIdx.x == 0) { stats[c] = ss[0]; stats[C + c] = sq[0]; }
}

// mode: 0 = bn, 1 = bn+relu, 2 = bn + add + relu
__global__ void k_bn_apply(const float* __restrict__ x, int ldx,
                           const float* __restrict__ stats,
                           const float* __restrict__ gam, const float* __restrict__ bet,
                           const float* __restrict__ add, int ldadd,
                           float* __restrict__ y, int ldy, int n, int C, int mode) {
  long idx = (long)blockIdx.x * blockDim.x + threadIdx.x;
  long tot = (long)n * C;
  if (idx >= tot) return;
  int c = (int)(idx % C);
  long i = idx / C;
  float inv_n = 1.f / (float)n;
  float mean = stats[c] * inv_n;
  float var  = stats[C + c] * inv_n - mean * mean;
  float v = (x[i * (long)ldx + c] - mean) * rsqrtf(var + 1e-5f) * gam[c] + bet[c];
  if (mode == 2) v += add[i * (long)ldadd + c];
  if (mode != 0) v = v > 0.f ? v : 0.f;
  y[i * (long)ldy + c] = v;
}

__global__ void k_scale(const float* __restrict__ x, const float* __restrict__ s,
                        float* __restrict__ y, long n, int C) {
  long idx = (long)blockIdx.x * blockDim.x + threadIdx.x;
  if (idx >= n * (long)C) return;
  int c = (int)(idx % C);
  y[idx] = x[idx] * s[c];
}

__global__ void k_copy_cols(const float* __restrict__ src, int Cs,
                            float* __restrict__ dst, int ldd, int coff, long n) {
  long idx = (long)blockIdx.x * blockDim.x + threadIdx.x;
  long tot = n * (long)Cs;
  if (idx >= tot) return;
  int c = (int)(idx % Cs);
  long i = idx / Cs;
  dst[i * (long)ldd + coff + c] = src[i * (long)Cs + c];
}

__global__ void k_bias_act(float* x, const float* __restrict__ b, long n, int C, int leaky) {
  long idx = (long)blockIdx.x * blockDim.x + threadIdx.x;
  if (idx >= n * (long)C) return;
  int c = (int)(idx % C);
  float v = x[idx] + b[c];
  if (leaky) v = v > 0.f ? v : 0.1f * v;
  x[idx] = v;
}

__global__ void k_embed(const float* __restrict__ t, const float* __restrict__ cond,
                        float* __restrict__ temb, float* __restrict__ cemb) {
  int j = threadIdx.x;
  if (j >= 48) return;
  float f = expf(-logf(10000.f) / 47.f * (float)j);
  float e = t[0] * f;
  temb[j] = sinf(e);
  temb[48 + j] = cosf(e);
  for (int k = 0; k < 8; ++k) {
    float ek = cond[k] * f;
    cemb[k * 96 + j] = sinf(ek);
    cemb[k * 96 + 48 + j] = cosf(ek);
  }
}

// FiLM for one stage (B=1): tiny MLPs, single block.
__global__ __launch_bounds__(256) void k_film(
    const float* __restrict__ temb, const float* __restrict__ cemb,
    const float* __restrict__ lW1, const float* __restrict__ lb1,
    const float* __restrict__ lW2, const float* __restrict__ lb2,
    const float* __restrict__ tW1, const float* __restrict__ tb1,
    const float* __restrict__ tW2, const float* __restrict__ tb2,
    const float* __restrict__ zW1, const float* __restrict__ zb1,
    const float* __restrict__ zW2, const float* __restrict__ zb2,
    int mid, int outC, float* __restrict__ sv)
{
  __shared__ float lh[768];   // latent hidden (8 x 96)
  __shared__ float th[96];    // temp hidden
  __shared__ float to2[256];  // temp out
  __shared__ float z[2048];   // (8 x 256) flattened
  __shared__ float hid[256];
  const int tid = threadIdx.x;

  for (int o = tid; o < 768; o += 256) {
    int k = o / 96, j = o % 96;
    float a = lb1[j];
    for (int i2 = 0; i2 < 96; ++i2) a += cemb[k * 96 + i2] * lW1[i2 * 96 + j];
    lh[o] = a > 0.f ? a : 0.1f * a;
  }
  for (int j = tid; j < 96; j += 256) {
    float a = tb1[j];
    for (int i2 = 0; i2 < 96; ++i2) a += temb[i2] * tW1[i2 * 96 + j];
    th[j] = a > 0.f ? a : 0.1f * a;
  }
  __syncthreads();
  for (int j = tid; j < 256; j += 256) {
    float a = tb2[j];
    for (int i2 = 0; i2 < 96; ++i2) a += th[i2] * tW2[i2 * 256 + j];
    to2[j] = a;
  }
  __syncthreads();
  for (int o = tid; o < 2048; o += 256) {
    int k = o / 256, j = o % 256;
    float a = lb2[j];
    for (int i2 = 0; i2 < 96; ++i2) a += lh[k * 96 + i2] * lW2[i2 * 256 + j];
    z[o] = a + to2[j];
  }
  __syncthreads();
  for (int m = tid; m < mid; m += 256) {
    float a = zb1[m];
    for (int i2 = 0; i2 < 2048; ++i2) a += z[i2] * zW1[i2 * mid + m];
    hid[m] = a > 0.f ? a : 0.1f * a;
  }
  __syncthreads();
  for (int o = tid; o < outC; o += 256) {
    float a = zb2[o];
    for (int i2 = 0; i2 < mid; ++i2) a += hid[i2] * zW2[i2 * outC + o];
    sv[o] = a;
  }
}

// ============================================================================
// Host-side orchestration
// ============================================================================
struct ResP  { const float *w1,*g1,*b1,*w2,*g2,*b2,*wd,*gd,*bd; bool has_wd; };
struct StageP{ const float *dw,*dg,*db; ResP r1, r2; };
struct UpP   { const float *uw,*ug,*ub; ResP r1, r2; };
struct CondP { const float *lW1,*lb1,*lW2,*lb2,*tW1,*tb1,*tW2,*tb2,*zW1,*zb1,*zW2,*zb2; };
struct Km    { const int* ins[27]; const int* outs[27]; int len[27]; int nk; };

// coverage modes for spconv
enum { COV_ACCUM = 0, COV_PARTITION = 1, COV_CENTER = 2 };

struct Arena {
  char* base; size_t off, cap;
  float* a(long n) {
    size_t b = ((size_t)n * 4 + 255) & ~(size_t)255;
    float* r = (float*)(base + off);
    off += b;
    return r;
  }
};

static inline void zero_buf(hipStream_t s, float* p, long n) {
  k_zero<<<dim3((unsigned)((n + 255) / 256)), dim3(256), 0, s>>>(p, n);
}

static void spconv(hipStream_t s, Arena& ar, const float* x, int ldx, const float* w,
                   int Cin, int Cout, const Km& km, float* y, int ldy, int cover) {
  const int KC = (Cin + 31) / 32, CT = (Cout + 15) / 16;
  // pack buffer: nk * KC * CT fragments * 1024 bytes = nk*KC*CT*256 floats
  unsigned short* pack = (unsigned short*)ar.a((long)km.nk * KC * CT * 256);
  k_pack_w<<<dim3(KC, CT, km.nk), dim3(32), 0, s>>>(w, Cin, Cout, pack);
  const size_t stride = (size_t)KC * CT * 512;   // ushorts per offset
  auto launch = [&](int k, int store) {
    int M = km.len[k];
    if (M <= 0) return;
    dim3 grid((M + 15) / 16, (Cout + 63) / 64);
    k_spconv_wmma<<<grid, dim3(32), 0, s>>>(x, ldx, pack + stride * k,
                                            km.ins[k], km.outs[k], M, y, ldy,
                                            Cin, Cout, CT, store);
  };
  if (cover == COV_CENTER) {
    launch(13, 1);
    for (int k = 0; k < km.nk; ++k) if (k != 13) launch(k, 0);
  } else {
    for (int k = 0; k < km.nk; ++k) launch(k, cover == COV_PARTITION ? 1 : 0);
  }
}

static void dense_mm(hipStream_t s, Arena& ar, const float* x, int ldx, const float* w,
                     int Cin, int Cout, int n, float* y, int ldy) {
  const int KC = (Cin + 31) / 32, CT = (Cout + 15) / 16;
  unsigned short* pack = (unsigned short*)ar.a((long)KC * CT * 256);
  k_pack_w<<<dim3(KC, CT, 1), dim3(32), 0, s>>>(w, Cin, Cout, pack);
  dim3 grid((n + 15) / 16, (Cout + 63) / 64);
  k_spconv_wmma<<<grid, dim3(32), 0, s>>>(x, ldx, pack, nullptr, nullptr, n, y, ldy,
                                          Cin, Cout, CT, 1);
}

static void bn(hipStream_t s, float* stats, const float* x, int ldx, int n, int C,
               const float* g, const float* b, const float* add, int ldadd,
               float* y, int ldy, int mode) {
  k_bn_stats<<<dim3(C), dim3(256), 0, s>>>(x, ldx, n, C, stats);
  long tot = (long)n * C;
  k_bn_apply<<<dim3((unsigned)((tot + 255) / 256)), dim3(256), 0, s>>>(
      x, ldx, stats, g, b, add, ldadd, y, ldy, n, C, mode);
}

static void res_block(hipStream_t s, const ResP& p, const Km& km,
                      const float* x, int n, int ci, int co,
                      float* out, Arena& ar, float* stats) {
  float* t1 = ar.a((long)n * co);
  float* t2 = ar.a((long)n * co);
  spconv(s, ar, x, ci, p.w1, ci, co, km, t1, co, COV_CENTER);
  bn(s, stats, t1, co, n, co, p.g1, p.b1, nullptr, 0, t1, co, 1);
  spconv(s, ar, t1, co, p.w2, co, co, km, t2, co, COV_CENTER);
  const float* addp; int ldadd;
  if (p.has_wd) {
    float* t3 = ar.a((long)n * co);
    dense_mm(s, ar, x, ci, p.wd, ci, co, n, t3, co);
    bn(s, stats, t3, co, n, co, p.gd, p.bd, nullptr, 0, t3, co, 0);
    addp = t3; ldadd = co;
  } else {
    addp = x; ldadd = ci;           // ci == co when no shortcut
  }
  bn(s, stats, t2, co, n, co, p.g2, p.b2, addp, ldadd, out, co, 2);
}

static void film(hipStream_t s, const CondP& c, const float* temb, const float* cemb,
                 int mid, int outC, float* sv) {
  k_film<<<dim3(1), dim3(256), 0, s>>>(temb, cemb,
      c.lW1, c.lb1, c.lW2, c.lb2, c.tW1, c.tb1, c.tW2, c.tb2,
      c.zW1, c.zb1, c.zW2, c.zb2, mid, outC, sv);
}

static void down_stage(hipStream_t s, const StageP& sp, const CondP& cp,
                       const Km& dnm, const Km& k3m,
                       const float* xin, int n_in, int ci, int co, int n_out,
                       const float* temb, const float* cemb, int mid,
                       float* out, Arena& ar, float* stats, float* filmbuf) {
  film(s, cp, temb, cemb, mid, ci, filmbuf);
  float* xs = ar.a((long)n_in * ci);
  long tin = (long)n_in * ci;
  k_scale<<<dim3((unsigned)((tin + 255) / 256)), dim3(256), 0, s>>>(xin, filmbuf, xs, n_in, ci);
  float* h = ar.a((long)n_out * ci);
  zero_buf(s, h, (long)n_out * ci);                      // many-to-one: accumulate
  spconv(s, ar, xs, ci, sp.dw, ci, ci, dnm, h, ci, COV_ACCUM);
  bn(s, stats, h, ci, n_out, ci, sp.dg, sp.db, nullptr, 0, h, ci, 1);
  float* r1o = ar.a((long)n_out * co);
  size_t mark = ar.off;
  res_block(s, sp.r1, k3m, h, n_out, ci, co, r1o, ar, stats);
  ar.off = mark;
  res_block(s, sp.r2, k3m, r1o, n_out, co, co, out, ar, stats);
}

static void up_stage(hipStream_t s, const UpP& up, const CondP& cp,
                     const Km& upm, const Km& k3m,
                     const float* xin, int n_in, int ci, int co, int csk,
                     const float* skip, int n_out,
                     const float* temb, const float* cemb, int mid,
                     float* out, Arena& ar, float* stats, float* filmbuf) {
  film(s, cp, temb, cemb, mid, ci, filmbuf);
  float* xs = ar.a((long)n_in * ci);
  long tin = (long)n_in * ci;
  k_scale<<<dim3((unsigned)((tin + 255) / 256)), dim3(256), 0, s>>>(xin, filmbuf, xs, n_in, ci);
  int wide = co + csk;
  float* cat = ar.a((long)n_out * wide);
  // up-conv offsets partition the fine rows: every row stored exactly once
  spconv(s, ar, xs, ci, up.uw, ci, co, upm, cat, wide, COV_PARTITION);
  bn(s, stats, cat, wide, n_out, co, up.ug, up.ub, nullptr, 0, cat, wide, 1);
  long tsk = (long)n_out * csk;
  k_copy_cols<<<dim3((unsigned)((tsk + 255) / 256)), dim3(256), 0, s>>>(skip, csk, cat, wide, co, n_out);
  float* r1o = ar.a((long)n_out * co);
  size_t mark = ar.off;
  res_block(s, up.r1, k3m, cat, n_out, wide, co, r1o, ar, stats);
  ar.off = mark;
  res_block(s, up.r2, k3m, r1o, n_out, co, co, out, ar, stats);
}

extern "C" void kernel_launch(void* const* d_in, const int* in_sizes, int n_in,
                              void* d_out, int out_size, void* d_ws, size_t ws_size,
                              hipStream_t stream) {
  (void)n_in; (void)out_size;
  int ii = 0;
  auto F = [&]() { return (const float*)d_in[ii++]; };
  auto I = [&](int& len) { len = in_sizes[ii]; const int* p = (const int*)d_in[ii]; ++ii; return p; };

  const float* feats   = F();
  const float* t_in    = F();
  const float* cond_in = F();

  auto readRes = [&](bool wd) {
    ResP r{};
    r.w1 = F(); r.g1 = F(); r.b1 = F(); r.w2 = F(); r.g2 = F(); r.b2 = F();
    r.has_wd = wd;
    if (wd) { r.wd = F(); r.gd = F(); r.bd = F(); }
    return r;
  };
  const float *stw1 = F(), *stg1 = F(), *stb1 = F(), *stw2 = F(), *stg2 = F(), *stb2 = F();
  auto readStage = [&](bool wd1) {
    StageP sp{}; sp.dw = F(); sp.dg = F(); sp.db = F();
    sp.r1 = readRes(wd1); sp.r2 = readRes(false); return sp;
  };
  StageP s1 = readStage(false), s2 = readStage(true), s3 = readStage(true), s4 = readStage(true);
  auto readUp = [&]() {
    UpP u{}; u.uw = F(); u.ug = F(); u.ub = F();
    u.r1 = readRes(true); u.r2 = readRes(false); return u;
  };
  UpP u1 = readUp(), u2 = readUp(), u3 = readUp(), u4 = readUp();
  auto readCond = [&]() {
    CondP c{};
    c.lW1 = F(); c.lb1 = F(); c.lW2 = F(); c.lb2 = F();
    c.tW1 = F(); c.tb1 = F(); c.tW2 = F(); c.tb2 = F();
    c.zW1 = F(); c.zb1 = F(); c.zW2 = F(); c.zb2 = F();
    return c;
  };
  CondP cs1 = readCond(), cs2 = readCond(), cs3 = readCond(), cs4 = readCond();
  CondP cu1 = readCond(), cu2 = readCond(), cu3 = readCond(), cu4 = readCond();
  const float *laW1 = F(), *lab1 = F(), *laW2 = F(), *lab2 = F();

  Km k3[5], dn[4], upm[4];
  for (int l = 0; l < 5; ++l) {
    k3[l].nk = 27;
    for (int k = 0; k < 27; ++k) k3[l].ins[k] = I(k3[l].len[k]);
    for (int k = 0; k < 27; ++k) { int d; k3[l].outs[k] = I(d); }
  }
  for (int l = 0; l < 4; ++l) {
    dn[l].nk = 8;
    for (int k = 0; k < 8; ++k) dn[l].ins[k] = I(dn[l].len[k]);
    for (int k = 0; k < 8; ++k) { int d; dn[l].outs[k] = I(d); }
  }
  for (int l = 0; l < 4; ++l) {            // transposed rulebook for up convs
    upm[l].nk = 8;
    for (int k = 0; k < 8; ++k) {
      upm[l].ins[k] = dn[l].outs[k];
      upm[l].outs[k] = dn[l].ins[k];
      upm[l].len[k] = dn[l].len[k];
    }
  }
  // point counts per level: center offset (index 13) of k3 is the identity map
  int ns[5];
  for (int l = 0; l < 5; ++l) ns[l] = k3[l].len[13];
  const int n0 = ns[0];

  Arena ar{(char*)d_ws, 0, ws_size};
  float* temb    = ar.a(96);
  float* cemb    = ar.a(8 * 96);
  float* filmbuf = ar.a(256);
  float* stats   = ar.a(512);
  float* x0 = ar.a((long)ns[0] * 32);
  float* x1 = ar.a((long)ns[1] * 32);
  float* x2 = ar.a((long)ns[2] * 64);
  float* x3 = ar.a((long)ns[3] * 128);
  float* x4 = ar.a((long)ns[4] * 256);
  float* y1 = ar.a((long)ns[3] * 256);
  float* y2 = ar.a((long)ns[2] * 128);
  float* y3 = ar.a((long)ns[1] * 96);
  float* y4 = ar.a((long)ns[0] * 96);
  const size_t smark = ar.off;             // scratch region, reset per stage

  k_embed<<<dim3(1), dim3(64), 0, stream>>>(t_in, cond_in, temb, cemb);

  // ---- stem ----
  {
    ar.off = smark;
    float* t1 = ar.a((long)n0 * 32);
    spconv(stream, ar, feats, 3, stw1, 3, 32, k3[0], t1, 32, COV_CENTER);
    bn(stream, stats, t1, 32, n0, 32, stg1, stb1, nullptr, 0, t1, 32, 1);
    spconv(stream, ar, t1, 32, stw2, 32, 32, k3[0], x0, 32, COV_CENTER);
    bn(stream, stats, x0, 32, n0, 32, stg2, stb2, nullptr, 0, x0, 32, 1);
  }
  // ---- encoder ----
  ar.off = smark;
  down_stage(stream, s1, cs1, dn[0], k3[1], x0, ns[0], 32, 32, ns[1], temb, cemb, 256, x1, ar, stats, filmbuf);
  ar.off = smark;
  down_stage(stream, s2, cs2, dn[1], k3[2], x1, ns[1], 32, 64, ns[2], temb, cemb, 256, x2, ar, stats, filmbuf);
  ar.off = smark;
  down_stage(stream, s3, cs3, dn[2], k3[3], x2, ns[2], 64, 128, ns[3], temb, cemb, 256, x3, ar, stats, filmbuf);
  ar.off = smark;
  down_stage(stream, s4, cs4, dn[3], k3[4], x3, ns[3], 128, 256, ns[4], temb, cemb, 256, x4, ar, stats, filmbuf);
  // ---- decoder ----
  ar.off = smark;
  up_stage(stream, u1, cu1, upm[3], k3[3], x4, ns[4], 256, 256, 128, x3, ns[3], temb, cemb, 256, y1, ar, stats, filmbuf);
  ar.off = smark;
  up_stage(stream, u2, cu2, upm[2], k3[2], y1, ns[3], 256, 128, 64, x2, ns[2], temb, cemb, 256, y2, ar, stats, filmbuf);
  ar.off = smark;
  up_stage(stream, u3, cu3, upm[1], k3[1], y2, ns[2], 128, 96, 32, x1, ns[1], temb, cemb, 128, y3, ar, stats, filmbuf);
  ar.off = smark;
  up_stage(stream, u4, cu4, upm[0], k3[0], y3, ns[1], 96, 96, 32, x0, ns[0], temb, cemb, 96, y4, ar, stats, filmbuf);

  // ---- final point-wise MLP: 96 -> 20 (leaky) -> 3 ----
  ar.off = smark;
  float* th = ar.a((long)n0 * 20);
  dense_mm(stream, ar, y4, 96, laW1, 96, 20, n0, th, 20);
  {
    long tot = (long)n0 * 20;
    k_bias_act<<<dim3((unsigned)((tot + 255) / 256)), dim3(256), 0, stream>>>(th, lab1, n0, 20, 1);
  }
  float* outp = (float*)d_out;
  dense_mm(stream, ar, th, 20, laW2, 20, 3, n0, outp, 3);
  {
    long tot = (long)n0 * 3;
    k_bias_act<<<dim3((unsigned)((tot + 255) / 256)), dim3(256), 0, stream>>>(outp, lab2, n0, 3, 0);
  }
}
